// Attention_1382979470038
// MI455X (gfx1250) — matrix operand order repaired and checked by reading, not compiled, not run on hardware
//
#include <hip/hip_runtime.h>
#include <hip/hip_bf16.h>
#include <stdint.h>

// MI455X / gfx1250: wave32, bf16 WMMA f32-accum for all GEMMs, flash-style
// attention (no materialized 4096x4096 score matrix -> saves ~768MB traffic).

typedef __bf16 bf16_t;
typedef __attribute__((ext_vector_type(16))) __bf16 v16bf;
typedef __attribute__((ext_vector_type(8)))  __bf16 v8bf;
typedef __attribute__((ext_vector_type(8)))  float  v8f;

union FragBF { v16bf v; v8bf h[2]; };

// A fragment (16x32 bf16, M x K): lane L -> row M = L%16;
// elems 0..7 : K = k0 + 8*(L>=16) + e ; elems 8..15 : K = k0 + 16 + 8*(L>=16) + (e-8)
__device__ __forceinline__ v16bf load_a_frag(const bf16_t* p, int ld, int row, int k0, int lane) {
    int r  = row + (lane & 15);
    int kk = k0 + ((lane >> 4) << 3);
    FragBF f;
    f.h[0] = *(const v8bf*)(p + (long)r * ld + kk);
    f.h[1] = *(const v8bf*)(p + (long)r * ld + kk + 16);
    return f.v;
}
// B fragment (32x16 bf16, K x N): lane L -> col N = L%16; elems 0..7 : K = k0 + 8*(L>=16) + e ; elems 8..15 : K = k0 + 16 + 8*(L>=16) + (e-8)
__device__ __forceinline__ v16bf load_b_frag(const bf16_t* p, int ld, int col, int k0, int lane) {
    int c  = col + (lane & 15);
    int kk = k0 + ((lane >> 4) << 3);
    FragBF f;
    f.h[0] = *(const v8bf*)(p + (long)c * ld + kk);
    f.h[1] = *(const v8bf*)(p + (long)c * ld + kk + 16);
    return f.v;
}
#define WMMA_BF16(a, b, c) \
    __builtin_amdgcn_wmma_f32_16x16x32_bf16(false, (a), false, (b), (short)0, (c), false, false)

// ---------------------------------------------------------------- stats ----
// One block per (b,c) row of 4096 pixels. Unbiased variance (ddof=1) + eps.
__global__ __launch_bounds__(256) void in_stats(
    const float* __restrict__ content, const float* __restrict__ style,
    float* __restrict__ meanC, float* __restrict__ rstdC,
    float* __restrict__ meanS, float* __restrict__ rstdS)
{
    const int row = blockIdx.x;                       // 0..2047 = b*512+c
    const float* p = (blockIdx.y ? style : content) + (long)row * 4096;
    __shared__ float rs[256], rq[256];
    float s = 0.f, q = 0.f;
    for (int i = threadIdx.x; i < 4096; i += 256) { float v = p[i]; s += v; q += v * v; }
    rs[threadIdx.x] = s; rq[threadIdx.x] = q;
    __syncthreads();
    for (int w = 128; w > 0; w >>= 1) {
        if ((int)threadIdx.x < w) { rs[threadIdx.x] += rs[threadIdx.x + w]; rq[threadIdx.x] += rq[threadIdx.x + w]; }
        __syncthreads();
    }
    if (threadIdx.x == 0) {
        float m   = rs[0] * (1.0f / 4096.0f);
        float var = (rq[0] - rs[0] * m) * (1.0f / 4095.0f);
        float rst = rsqrtf(var + 1e-5f);
        if (blockIdx.y) { meanS[row] = m; rstdS[row] = rst; }
        else            { meanC[row] = m; rstdC[row] = rst; }
    }
}

// -------------------------------------------------- norm + transpose -------
// [b][c][n] f32 -> [b][n][c] bf16 (normalized) and optionally raw bf16 too.
__global__ __launch_bounds__(256) void norm_tr(
    const float* __restrict__ src, const float* __restrict__ mean,
    const float* __restrict__ rstd, bf16_t* __restrict__ dstN,
    bf16_t* __restrict__ dstRaw)
{
    __shared__ float t[32][33];
    const int tx = threadIdx.x, ty = threadIdx.y;          // blockDim (32,8)
    const int c0 = blockIdx.x * 32, n0 = blockIdx.y * 32, b = blockIdx.z;
    const long sOff = (long)b * 512 * 4096;
    for (int i = 0; i < 4; ++i) {
        int c = c0 + ty + 8 * i;
        t[ty + 8 * i][tx] = src[sOff + (long)c * 4096 + n0 + tx];
    }
    __syncthreads();
    const long dOff = (long)b * 4096 * 512;
    for (int i = 0; i < 4; ++i) {
        int n = n0 + ty + 8 * i;
        int c = c0 + tx;
        float v = t[tx][ty + 8 * i];
        long idx = dOff + (long)n * 512 + c;
        dstN[idx] = (bf16_t)((v - mean[b * 512 + c]) * rstd[b * 512 + c]);
        if (dstRaw) dstRaw[idx] = (bf16_t)v;
    }
}

// ------------------------------------------------------- weight cvt --------
__global__ void cvt_weights(const float* __restrict__ f, const float* __restrict__ g,
                            const float* __restrict__ h, const float* __restrict__ o,
                            bf16_t* __restrict__ dst)
{
    int i = blockIdx.x * 256 + threadIdx.x;                // 0..262143
    dst[i]              = (bf16_t)f[i];
    dst[262144 + i]     = (bf16_t)g[i];
    dst[2 * 262144 + i] = (bf16_t)h[i];
    dst[3 * 262144 + i] = (bf16_t)o[i];
}

// --------------------------------------------------------- NT GEMM ---------
// D[i][j] = sum_k A[i][k]*B[j][k] + bias   (A:[M][K] bf16, B:[N][K] bf16)
// 128x128 block tile, 8 waves in 2x4 grid, each wave 64x32 via 4x2 WMMA accums.
__global__ __launch_bounds__(256) void gemm_nt(
    const bf16_t* __restrict__ A, long aStride,
    const bf16_t* __restrict__ B, long bStride,
    const float* __restrict__ bias, int biasRow,
    void* __restrict__ Dv, int ldd, long dStride, int outF32, int Kdim)
{
    const int tid  = threadIdx.x;
    const int lane = tid & 31;
    const int wave = tid >> 5;
    const int wr = wave >> 2;                  // 0..1
    const int wc = wave & 3;                   // 0..3
    const int i0 = blockIdx.y * 128 + wr * 64;
    const int j0 = blockIdx.x * 128 + wc * 32;
    const int bz = blockIdx.z;
    const bf16_t* Ab = A + (long)bz * aStride;
    const bf16_t* Bb = B + (long)bz * bStride;

    const v8f vz = {0.f, 0.f, 0.f, 0.f, 0.f, 0.f, 0.f, 0.f};
    v8f acc[4][2];
    for (int r = 0; r < 4; ++r) for (int c = 0; c < 2; ++c) acc[r][c] = vz;

    for (int k0 = 0; k0 < Kdim; k0 += 32) {
        v16bf b0 = load_b_frag(Bb, Kdim, j0,      k0, lane);
        v16bf b1 = load_b_frag(Bb, Kdim, j0 + 16, k0, lane);
        for (int r = 0; r < 4; ++r) {
            v16bf a = load_a_frag(Ab, Kdim, i0 + r * 16, k0, lane);
            acc[r][0] = WMMA_BF16(a, b0, acc[r][0]);
            acc[r][1] = WMMA_BF16(a, b1, acc[r][1]);
        }
    }

    for (int r = 0; r < 4; ++r) {
        const int rbase = i0 + r * 16 + ((lane >> 4) << 3);
        for (int v = 0; v < 8; ++v) {
            const int i = rbase + v;
            for (int c = 0; c < 2; ++c) {
                const int j = j0 + c * 16 + (lane & 15);
                float val = acc[r][c][v] + (biasRow ? bias[i] : bias[j]);
                if (outF32) ((float*)Dv)[(long)bz * dStride + (long)i * ldd + j] = val;
                else        ((bf16_t*)Dv)[(long)bz * dStride + (long)i * ldd + j] = (bf16_t)val;
            }
        }
    }
}

// --------------------------------------------------- flash attention -------
// Q,K: [b][4096][512] bf16 ; V: [b][512][4096] bf16 ; out styledT: [b][4096][512] bf16
// Block: 32 query rows, 8 waves. Per iter: 128 keys. Wave w computes S slab
// [32 x 16] (keys w*16..), then owns channel slice c in [w*64, w*64+64) for P*V.
__global__ __launch_bounds__(256) void attn_kernel(
    const bf16_t* __restrict__ Q, const bf16_t* __restrict__ Km,
    const bf16_t* __restrict__ V, bf16_t* __restrict__ styledT)
{
    const int tid  = threadIdx.x;
    const int lane = tid & 31;
    const int wave = tid >> 5;
    const int n0   = blockIdx.x * 32;
    const int b    = blockIdx.y;
    const long qkOff = (long)b * 4096 * 512;
    const bf16_t* Qb = Q  + qkOff;
    const bf16_t* Kb = Km + qkOff;
    const bf16_t* Vb = V  + (long)b * 512 * 4096;
    bf16_t*       Ob = styledT + qkOff;

    __shared__ bf16_t sQ[32 * 512];     // 32 KB query tile
    __shared__ float  sS[32 * 128];     // 16 KB score slab
    __shared__ bf16_t sP[32 * 128];     //  8 KB exp(S-max) slab
    __shared__ float  sMax[32], sSum[32], sAlpha[32];

    {   // stage Q tile (contiguous 32KB) into LDS
        const uint4* src = (const uint4*)(Qb + (long)n0 * 512);
        uint4* dst = (uint4*)sQ;
        for (int i = tid; i < 32 * 512 * 2 / 16; i += 256) dst[i] = src[i];
    }
    if (tid < 32) { sMax[tid] = -3.0e38f; sSum[tid] = 0.f; }
    __syncthreads();

    const v8f vz = {0.f, 0.f, 0.f, 0.f, 0.f, 0.f, 0.f, 0.f};
    v8f acc[2][4];
    for (int nt = 0; nt < 2; ++nt) for (int ct = 0; ct < 4; ++ct) acc[nt][ct] = vz;

    for (int it = 0; it < 32; ++it) {
        const int m0 = it * 128;
        if (it + 1 < 32)    // warm L2/WGP$ for next key tile
            __builtin_prefetch(Kb + (long)(m0 + 128 + wave * 16 + (lane & 15)) * 512, 0, 0);

        // ---- S slab: rows 0..31, keys m0 + wave*16 .. +16, full 512-dim dot
        for (int nt = 0; nt < 2; ++nt) {
            v8f s = vz;
            for (int kc = 0; kc < 16; ++kc) {
                v16bf a = load_a_frag(sQ, 512, nt * 16, kc * 32, lane);
                v16bf bb = load_b_frag(Kb, 512, m0 + wave * 16, kc * 32, lane);
                s = WMMA_BF16(a, bb, s);
            }
            const int jj = wave * 16 + (lane & 15);
            const int rbase = nt * 16 + ((lane >> 4) << 3);
            for (int v = 0; v < 8; ++v) sS[(rbase + v) * 128 + jj] = s[v];
        }
        __syncthreads();

        // ---- online softmax stats (one thread per row)
        if (tid < 32) {
            const float* row = &sS[tid * 128];
            float cm = -3.0e38f;
            for (int m = 0; m < 128; ++m) cm = fmaxf(cm, row[m]);
            float om = sMax[tid];
            float nm = fmaxf(om, cm);
            float al = __expf(om - nm);
            float s = 0.f;
            for (int m = 0; m < 128; ++m) s += __expf(row[m] - nm);
            sSum[tid] = sSum[tid] * al + s;
            sMax[tid] = nm;
            sAlpha[tid] = al;
        }
        __syncthreads();

        // ---- P = exp(S - rowmax) as bf16 (256 threads x 16 values)
        {
            int r  = tid >> 3;
            int mb = (tid & 7) * 16;
            float nm = sMax[r];
            for (int q = 0; q < 16; ++q)
                sP[r * 128 + mb + q] = (bf16_t)__expf(sS[r * 128 + mb + q] - nm);
        }
        __syncthreads();

        // ---- rescale accumulators by alpha[row]
        for (int nt = 0; nt < 2; ++nt) {
            const int rbase = nt * 16 + ((lane >> 4) << 3);
            for (int v = 0; v < 8; ++v) {
                float al = sAlpha[rbase + v];
                for (int ct = 0; ct < 4; ++ct) acc[nt][ct][v] *= al;
            }
        }

        // ---- acc += P * V^T over this wave's 64-channel slice
        for (int ks = 0; ks < 4; ++ks) {
            v16bf aP0 = load_a_frag(sP, 128, 0,  ks * 32, lane);
            v16bf aP1 = load_a_frag(sP, 128, 16, ks * 32, lane);
            for (int ct = 0; ct < 4; ++ct) {
                v16bf bb = load_b_frag(Vb, 4096, wave * 64 + ct * 16, m0 + ks * 32, lane);
                acc[0][ct] = WMMA_BF16(aP0, bb, acc[0][ct]);
                acc[1][ct] = WMMA_BF16(aP1, bb, acc[1][ct]);
            }
        }
        __syncthreads();
    }

    // ---- epilogue: divide by softmax denominator, store styledT[n][c]
    for (int nt = 0; nt < 2; ++nt) {
        const int rbase = nt * 16 + ((lane >> 4) << 3);
        for (int v = 0; v < 8; ++v) {
            float inv = 1.0f / sSum[rbase + v];
            int n = n0 + rbase + v;
            for (int ct = 0; ct < 4; ++ct) {
                int c = wave * 64 + ct * 16 + (lane & 15);
                Ob[(long)n * 512 + c] = (bf16_t)(acc[nt][ct][v] * inv);
            }
        }
    }
}

// ----------------------------------------------------------- launcher ------
extern "C" void kernel_launch(void* const* d_in, const int* in_sizes, int n_in,
                              void* d_out, int out_size, void* d_ws, size_t ws_size,
                              hipStream_t stream)
{
    const float* content = (const float*)d_in[0];
    const float* style   = (const float*)d_in[1];
    const float* f_w = (const float*)d_in[2];
    const float* f_b = (const float*)d_in[3];
    const float* g_w = (const float*)d_in[4];
    const float* g_b = (const float*)d_in[5];
    const float* h_w = (const float*)d_in[6];
    const float* h_b = (const float*)d_in[7];
    const float* o_w = (const float*)d_in[8];
    const float* o_b = (const float*)d_in[9];

    char* ws = (char*)d_ws;
    size_t off = 0;
    auto alloc = [&](size_t bytes) -> char* {
        char* p = ws + off;
        off += (bytes + 255) & ~(size_t)255;
        return p;
    };
    float*  meanC = (float*)alloc(2048 * 4);
    float*  rstdC = (float*)alloc(2048 * 4);
    float*  meanS = (float*)alloc(2048 * 4);
    float*  rstdS = (float*)alloc(2048 * 4);
    bf16_t* wbf   = (bf16_t*)alloc((size_t)4 * 262144 * 2);
    const size_t BIG = (size_t)4 * 4096 * 512 * 2;     // 16 MB each
    bf16_t* Xc  = (bf16_t*)alloc(BIG);                 // normed content [b][n][c]
    bf16_t* Xs  = (bf16_t*)alloc(BIG);                 // normed style   [b][n][c]
    bf16_t* Xsr = (bf16_t*)alloc(BIG);                 // raw style      [b][n][c]
    bf16_t* Qm  = (bf16_t*)alloc(BIG);                 // F^T [b][n][o]
    bf16_t* Kmm = (bf16_t*)alloc(BIG);                 // G^T [b][m][o]
    bf16_t* Vm  = (bf16_t*)alloc(BIG);                 // H   [b][o][m]
    bf16_t* styledT = Xc;                              // reuse: Xc dead after Q proj

    const long XS = (long)4096 * 512;                  // per-batch elements

    in_stats<<<dim3(2048, 2), 256, 0, stream>>>(content, style, meanC, rstdC, meanS, rstdS);
    cvt_weights<<<1024, 256, 0, stream>>>(f_w, g_w, h_w, o_w, wbf);
    norm_tr<<<dim3(16, 128, 4), dim3(32, 8), 0, stream>>>(content, meanC, rstdC, Xc, nullptr);
    norm_tr<<<dim3(16, 128, 4), dim3(32, 8), 0, stream>>>(style,   meanS, rstdS, Xs, Xsr);

    // Q = Xc * f_w^T + f_b      [4096 x 512] per batch (bias over j=o)
    gemm_nt<<<dim3(4, 32, 4), 256, 0, stream>>>(Xc, XS, wbf + 0 * 262144, 0,
                                                f_b, 0, Qm, 512, XS, 0, 512);
    // K = Xs * g_w^T + g_b
    gemm_nt<<<dim3(4, 32, 4), 256, 0, stream>>>(Xs, XS, wbf + 1 * 262144, 0,
                                                g_b, 0, Kmm, 512, XS, 0, 512);
    // V = h_w * Xsr^T + h_b     [512 x 4096] per batch (bias over i=o)
    gemm_nt<<<dim3(32, 4, 4), 256, 0, stream>>>(wbf + 2 * 262144, 0, Xsr, XS,
                                                h_b, 1, Vm, 4096, XS, 0, 512);

    attn_kernel<<<dim3(128, 4), 256, 0, stream>>>(Qm, Kmm, Vm, styledT);

    // out = o_w * styledT^T + o_b   f32 [512 x 4096] per batch -> d_out
    gemm_nt<<<dim3(32, 4, 4), 256, 0, stream>>>(wbf + 3 * 262144, 0, styledT, XS,
                                                o_b, 1, (float*)d_out, 4096, XS, 1, 512);

    (void)in_sizes; (void)n_in; (void)out_size; (void)ws_size;
}
